// LUTModule_1005022347825
// MI455X (gfx1250) — matrix operand order repaired and checked by reading, not compiled, not run on hardware
//
#include <hip/hip_runtime.h>
#include <math.h>

typedef __attribute__((ext_vector_type(16))) __bf16 v16bf;
typedef __attribute__((ext_vector_type(8)))  __bf16 v8bf;
typedef __attribute__((ext_vector_type(8)))  float  v8f;
typedef __attribute__((ext_vector_type(4)))  int    v4i;

#define HH 384
#define WWID 384
#define HWSZ (HH*WWID)
#define BB 4
#define GG 33
#define G3C (GG*GG*GG)          /* 35937 */
#define LLUT 8
#define NBIG (LLUT*3*G3C)       /* 862488 */

__device__ __forceinline__ float gelu_f(float x) {
    return 0.5f * x * (1.0f + erff(x * 0.70710678118654752f));
}

// ---------------------------------------------------------------- embedding mean
__global__ void k_embed_mean(const int* __restrict__ tok,
                             const float* __restrict__ emb,
                             float* __restrict__ tf) {
    int b = blockIdx.x, e = threadIdx.x;
    float s = 0.f;
    for (int t = 0; t < 1024; ++t)
        s += emb[(size_t)tok[b*1024 + t] * 256 + e];
    tf[b*256 + e] = s * (1.0f/1024.0f);
}

// ---------------------------------------------------------------- small dense (B=4 rows)
__global__ void k_dense(const float* __restrict__ in, const float* __restrict__ w,
                        const float* __restrict__ bias, float* __restrict__ out,
                        int K, int N, int act) {
    __shared__ float row[256];
    int b = blockIdx.x, tid = threadIdx.x;
    if (tid < K) row[tid] = in[b*K + tid];
    __syncthreads();
    if (tid < N) {
        float acc = bias[tid];
        for (int k = 0; k < K; ++k) acc += row[k] * w[(size_t)k*N + tid];
        out[b*N + tid] = act ? gelu_f(acc) : acc;
    }
}

// ---------------------------------------------------------------- softmax over L=8
__global__ void k_softmax8(const float* __restrict__ in, float* __restrict__ out) {
    int b = threadIdx.x;
    if (b >= BB) return;
    float m = -1e30f;
    for (int l = 0; l < LLUT; ++l) m = fmaxf(m, in[b*LLUT + l]);
    float e[LLUT]; float s = 0.f;
    for (int l = 0; l < LLUT; ++l) { e[l] = __expf(in[b*LLUT + l] - m); s += e[l]; }
    float inv = 1.0f / s;
    for (int l = 0; l < LLUT; ++l) out[b*LLUT + l] = e[l] * inv;
}

// ---------------------------------------------------------------- big GEMM fused with LUT mixing
// lut_fused[b][c][i] = sum_l lw[b,l] * ( h[b,:] . W[:, (l*3+c)*G3+i] + b2[...] )
// 4 consecutive i per thread (scalar-unrolled: G3 is odd, vector alignment impossible)
__global__ void k_lut_gemm(const float* __restrict__ h, const float* __restrict__ w2,
                           const float* __restrict__ b2, const float* __restrict__ lw,
                           float* __restrict__ lutf) {
    __shared__ float sh[BB][256];
    __shared__ float slw[BB][LLUT];
    int tid = threadIdx.x;
    for (int e = tid; e < BB*256; e += 256) sh[e >> 8][e & 255] = h[e];
    if (tid < BB*LLUT) slw[tid / LLUT][tid % LLUT] = lw[tid];
    __syncthreads();
    int i0 = (blockIdx.x * 256 + tid) * 4;
    if (i0 >= G3C) return;
    const int nv = (G3C - i0 < 4) ? (G3C - i0) : 4;
    float fo[BB][3][4] = {};
    for (int l = 0; l < LLUT; ++l) {
        for (int c = 0; c < 3; ++c) {
            size_t base = (size_t)(l*3 + c) * G3C + i0;
            float acc[BB][4] = {};
            for (int k = 0; k < 256; ++k) {
                const float* wp = &w2[(size_t)k*NBIG + base];
                if ((k & 63) == 0)
                    __builtin_prefetch(wp + (size_t)64*NBIG, 0, 1);
                #pragma unroll
                for (int j = 0; j < 4; ++j) {
                    if (j < nv) {
                        float wv = wp[j];
                        acc[0][j] += sh[0][k] * wv;
                        acc[1][j] += sh[1][k] * wv;
                        acc[2][j] += sh[2][k] * wv;
                        acc[3][j] += sh[3][k] * wv;
                    }
                }
            }
            #pragma unroll
            for (int j = 0; j < 4; ++j) {
                if (j < nv) {
                    float bb = b2[base + j];
                    #pragma unroll
                    for (int b = 0; b < BB; ++b)
                        fo[b][c][j] += slw[b][l] * (acc[b][j] + bb);
                }
            }
        }
    }
    for (int b = 0; b < BB; ++b)
        for (int c = 0; c < 3; ++c)
            for (int j = 0; j < 4; ++j)
                if (j < nv)
                    lutf[((size_t)b*3 + c) * G3C + i0 + j] = fo[b][c][j];
}

// ---------------------------------------------------------------- trilinear LUT apply -> comb ch 0..2
__global__ void k_apply_lut(const float* __restrict__ lutf,
                            const float* __restrict__ recon,
                            float* __restrict__ comb) {
    int idx = blockIdx.x * 256 + threadIdx.x;
    if (idx >= BB*HWSZ) return;
    int b = idx / HWSZ, p = idx - b*HWSZ;
    float cr = fminf(fmaxf(recon[((size_t)b*3 + 0)*HWSZ + p] * 32.f, 0.f), 32.f);
    float cg = fminf(fmaxf(recon[((size_t)b*3 + 1)*HWSZ + p] * 32.f, 0.f), 32.f);
    float cb = fminf(fmaxf(recon[((size_t)b*3 + 2)*HWSZ + p] * 32.f, 0.f), 32.f);
    float fr = fminf(floorf(cr), 31.f);
    float fg = fminf(floorf(cg), 31.f);
    float fb = fminf(floorf(cb), 31.f);
    int r0 = (int)fr, g0 = (int)fg, b0 = (int)fb;
    float tr = cr - fr, tg = cg - fg, tb = cb - fb;
    float w000 = (1.f-tr)*(1.f-tg)*(1.f-tb), w001 = (1.f-tr)*(1.f-tg)*tb;
    float w010 = (1.f-tr)*tg*(1.f-tb),       w011 = (1.f-tr)*tg*tb;
    float w100 = tr*(1.f-tg)*(1.f-tb),       w101 = tr*(1.f-tg)*tb;
    float w110 = tr*tg*(1.f-tb),             w111 = tr*tg*tb;
    int base = (r0*GG + g0)*GG + b0;
    for (int c = 0; c < 3; ++c) {
        const float* L = lutf + ((size_t)b*3 + c)*G3C;
        float acc = L[base            ]*w000 + L[base + 1          ]*w001
                  + L[base + GG       ]*w010 + L[base + GG + 1     ]*w011
                  + L[base + GG*GG    ]*w100 + L[base + GG*GG + 1  ]*w101
                  + L[base + GG*GG+GG ]*w110 + L[base + GG*GG+GG+1 ]*w111;
        comb[((size_t)b*6 + c)*HWSZ + p] = acc;
    }
}

// ---------------------------------------------------------------- weight pack: OIHW fp32 -> [tap][n=co(32)][k=ci(32)] bf16, BN folded
__global__ void k_pack(const float* __restrict__ w, const float* __restrict__ bias,
                       const float* __restrict__ g, const float* __restrict__ bt,
                       int Cout, int Cin, int use_bn,
                       __bf16* __restrict__ wpk, float* __restrict__ bpk) {
    const float RS = 0.99999500003749969f; // (1+1e-5)^-0.5
    int idx = blockIdx.x * 256 + threadIdx.x;
    if (idx < 9*32*32) {
        int t = idx / 1024, r = idx % 1024, n = r / 32, k = r % 32;
        float v = 0.f;
        if (k < Cin && n < Cout) {
            v = w[((size_t)n*Cin + k)*9 + t];
            if (use_bn) v *= g[n] * RS;
        }
        wpk[idx] = (__bf16)v;   // [t][n][k], k contiguous -> B fragment = 1x 32B LDS read
    }
    if (idx < 32) {
        float bv = 0.f;
        if (idx < Cout) {
            bv = bias[idx];
            if (use_bn) bv = bv * (g[idx] * RS) + bt[idx];
        }
        bpk[idx] = bv;
    }
}

// ---------------------------------------------------------------- WMMA implicit-GEMM 3x3 conv
// tile: M=16 pixels (row segment), N=32 out-channels (2 accumulators), K=9 taps x 32 in-ch
// flags: 1 = gelu, 2 = add skip before gelu
__global__ __launch_bounds__(128)
void k_conv_wmma(const float* __restrict__ in, int inCtot, int inC0, int Cin,
                 const __bf16* __restrict__ wpk, const float* __restrict__ bpk,
                 float* __restrict__ out, int outCtot, int outC0, int Cout,
                 const float* __restrict__ skip, int skipCtot, int skipC0, int flags) {
    __shared__ __bf16 sW[9][32][32];          // [tap][n][k] : k contiguous
    __shared__ float  sB[32];
    __shared__ __bf16 sA[4][3][18][32];       // per-wave [dy][x][ci] : ci contiguous
    __shared__ float  sOut[4][32][16];        // per-wave [n][x] store-transpose tile

    const int tid  = threadIdx.x;
    const int wave = tid >> 5;
    const int lane = tid & 31;

    // stage packed weights: 18432B global -> LDS
#if __has_builtin(__builtin_amdgcn_global_load_async_to_lds_b128)
    {
        // gfx1250 async copy engine: no VGPR round-trip, tracked by ASYNCcnt
        typedef __attribute__((address_space(1))) v4i gv4i;
        typedef __attribute__((address_space(3))) v4i lv4i;
        gv4i* g = (gv4i*)wpk;
        lv4i* l = (lv4i*)sW;
        #pragma unroll
        for (int e = 0; e < 9; ++e)
            __builtin_amdgcn_global_load_async_to_lds_b128(
                g + (e*128 + tid), l + (e*128 + tid), 0, 0);
#if __has_builtin(__builtin_amdgcn_s_wait_asynccnt)
        __builtin_amdgcn_s_wait_asynccnt(0);
#else
        asm volatile("s_wait_asynccnt 0x0" ::: "memory");
#endif
    }
#else
    {
        const uint4* src = (const uint4*)wpk;
        uint4* dst = (uint4*)sW;
        #pragma unroll
        for (int e = 0; e < 9; ++e) dst[e*128 + tid] = src[e*128 + tid];
    }
#endif
    if (tid < 32) sB[tid] = bpk[tid];

    const int TX   = WWID / 16;
    const int tile = blockIdx.x * 4 + wave;
    const int b    = tile / (HH*TX);
    const int rm   = tile % (HH*TX);
    const int y    = rm / TX;
    const int x0   = (rm % TX) * 16;

    // stage input window (x0-1..x0+16, y-1..y+1, 32 ch zero-padded) as bf16
    // global reads x-contiguous (coalesced); LDS writes channel-transposed
    for (int e = lane; e < 3*32*18; e += 32) {
        int dy = e / (32*18);
        int r2 = e % (32*18);
        int ci = r2 / 18;
        int xx = r2 % 18;
        int gx = x0 - 1 + xx;
        int gy = y - 1 + dy;
        float v = 0.f;
        if (ci < Cin && gx >= 0 && gx < WWID && gy >= 0 && gy < HH)
            v = in[((size_t)(b*inCtot + inC0 + ci))*HWSZ + (size_t)gy*WWID + gx];
        sA[wave][dy][xx][ci] = (__bf16)v;
    }
    __syncthreads();

    const int half = lane >> 4;
    const int m    = lane & 15;
    v8f acc0 = {};
    v8f acc1 = {};

    #pragma unroll
    for (int t = 0; t < 9; ++t) {
        const int dy = t / 3, dxi = t % 3;
        // A fragment (16x32 bf16): elements 0..15 = K base..base+15 contiguous
        // lane<16: K 0..15; lane>=16: K 16..31
        const v8bf* arow = (const v8bf*)&sA[wave][dy][dxi + m][0];
        v8bf alo = arow[2*half];
        v8bf ahi = arow[2*half + 1];
        v16bf a = __builtin_shufflevector(alo, ahi,
                    0,1,2,3,4,5,6,7,8,9,10,11,12,13,14,15);
        // B fragments: lane<16 -> K 0..15, lane>=16 -> K 16..31, col n
        v16bf bf0 = *(const v16bf*)&sW[t][m][16*half];
        acc0 = __builtin_amdgcn_wmma_f32_16x16x32_bf16(false, a, false, bf0,
                                                       (short)0, acc0, false, false);
        if (Cout > 16) {
            v16bf bf1 = *(const v16bf*)&sW[t][16 + m][16*half];
            acc1 = __builtin_amdgcn_wmma_f32_16x16x32_bf16(false, a, false, bf1,
                                                           (short)0, acc1, false, false);
        }
    }

    // ---- epilogue: transpose through LDS so global stores are x-contiguous
    // D layout: lane<16: M=r, N=lane ; lane>=16: M=8+r, N=lane-16
    {
        float4 lo0 = make_float4(acc0[0], acc0[1], acc0[2], acc0[3]);
        float4 hi0 = make_float4(acc0[4], acc0[5], acc0[6], acc0[7]);
        *(float4*)&sOut[wave][m][8*half]     = lo0;
        *(float4*)&sOut[wave][m][8*half + 4] = hi0;
    }
    if (Cout > 16) {
        float4 lo1 = make_float4(acc1[0], acc1[1], acc1[2], acc1[3]);
        float4 hi1 = make_float4(acc1[4], acc1[5], acc1[6], acc1[7]);
        *(float4*)&sOut[wave][16 + m][8*half]     = lo1;
        *(float4*)&sOut[wave][16 + m][8*half + 4] = hi1;
    }
    // same-wave LDS ordering is in-order; compiler inserts dscnt waits
    for (int id = lane; id < Cout*4; id += 32) {
        int n  = id >> 2;
        int xq = (id & 3) * 4;
        float4 v = *(const float4*)&sOut[wave][n][xq];
        float bb = sB[n];
        v.x += bb; v.y += bb; v.z += bb; v.w += bb;
        if (flags & 2) {
            float4 sk = *(const float4*)&skip[((size_t)(b*skipCtot + skipC0 + n))*HWSZ
                                              + (size_t)y*WWID + x0 + xq];
            v.x += sk.x; v.y += sk.y; v.z += sk.z; v.w += sk.w;
        }
        if (flags & 1) {
            v.x = gelu_f(v.x); v.y = gelu_f(v.y); v.z = gelu_f(v.z); v.w = gelu_f(v.w);
        }
        *(float4*)&out[((size_t)(b*outCtot + outC0 + n))*HWSZ
                       + (size_t)y*WWID + x0 + xq] = v;
    }
}

// ---------------------------------------------------------------- workspace layout
static constexpr size_t OFF_TF    = 0;
static constexpr size_t OFF_HA    = 4096;
static constexpr size_t OFF_TF2   = 8192;
static constexpr size_t OFF_HL    = 12288;
static constexpr size_t OFF_HWG   = 16384;
static constexpr size_t OFF_LWRAW = 17408;
static constexpr size_t OFF_LW    = 17664;
static constexpr size_t OFF_WPK   = 20480;       // 8 layers * 20480 (18432 bf16 + 128 bias + pad)
static constexpr size_t WPK_STRIDE= 20480;
static constexpr size_t OFF_LUTF  = 262144;      // 4*3*35937*4 = 1,724,976
static constexpr size_t OFF_COMB  = 2097152;     // 4*6*HW*4 = 14,155,776
static constexpr size_t OFF_BUFA  = 17825792;    // 4*32*HW*4 = 75,497,472 each
static constexpr size_t OFF_BUFB  = OFF_BUFA + 75497472ull;
static constexpr size_t OFF_BUFC  = OFF_BUFB + 75497472ull;

extern "C" void kernel_launch(void* const* d_in, const int* in_sizes, int n_in,
                              void* d_out, int out_size, void* d_ws, size_t ws_size,
                              hipStream_t stream) {
    (void)in_sizes; (void)n_in; (void)out_size; (void)ws_size;
    const int*   tokens  = (const int*)  d_in[0];
    const float* img     = (const float*)d_in[1];
    const float* recon   = (const float*)d_in[2];
    const float* emb     = (const float*)d_in[3];
    const float* agg_w1  = (const float*)d_in[4];
    const float* agg_b1  = (const float*)d_in[5];
    const float* agg_w2  = (const float*)d_in[6];
    const float* agg_b2  = (const float*)d_in[7];
    const float* lut_w1  = (const float*)d_in[8];
    const float* lut_b1  = (const float*)d_in[9];
    const float* lut_w2  = (const float*)d_in[10];
    const float* lut_b2  = (const float*)d_in[11];
    const float* wg_w1   = (const float*)d_in[12];
    const float* wg_b1   = (const float*)d_in[13];
    const float* wg_w2   = (const float*)d_in[14];
    const float* wg_b2   = (const float*)d_in[15];

    char* ws = (char*)d_ws;
    float* tf    = (float*)(ws + OFF_TF);
    float* hA    = (float*)(ws + OFF_HA);
    float* tf2   = (float*)(ws + OFF_TF2);
    float* hL    = (float*)(ws + OFF_HL);
    float* hWg   = (float*)(ws + OFF_HWG);
    float* lwraw = (float*)(ws + OFF_LWRAW);
    float* lw    = (float*)(ws + OFF_LW);
    float* lutf  = (float*)(ws + OFF_LUTF);
    float* comb  = (float*)(ws + OFF_COMB);
    float* bufA  = (float*)(ws + OFF_BUFA);
    float* bufB  = (float*)(ws + OFF_BUFB);
    float* bufC  = (float*)(ws + OFF_BUFC);
    float* outp  = (float*)d_out;

    auto wpk = [&](int i) { return (__bf16*)(ws + OFF_WPK + (size_t)i*WPK_STRIDE); };
    auto bpk = [&](int i) { return (float*)(ws + OFF_WPK + (size_t)i*WPK_STRIDE + 18432); };

    // ---- token path
    k_embed_mean<<<BB, 256, 0, stream>>>(tokens, emb, tf);
    k_dense<<<BB, 256, 0, stream>>>(tf,  agg_w1, agg_b1, hA,    256, 256, 1);
    k_dense<<<BB, 256, 0, stream>>>(hA,  agg_w2, agg_b2, tf2,   256, 256, 0);
    k_dense<<<BB, 256, 0, stream>>>(tf2, lut_w1, lut_b1, hL,    256, 256, 1);
    k_dense<<<BB, 256, 0, stream>>>(tf2, wg_w1,  wg_b1,  hWg,   256, 64,  1);
    k_dense<<<BB, 256, 0, stream>>>(hWg, wg_w2,  wg_b2,  lwraw, 64,  8,   0);
    k_softmax8<<<1, BB, 0, stream>>>(lwraw, lw);

    // ---- memory-bound mega-GEMM fused with softmax LUT mixing (4 i / thread)
    k_lut_gemm<<<(G3C + 1023)/1024, 256, 0, stream>>>(hL, lut_w2, lut_b2, lw, lutf);
    k_apply_lut<<<(BB*HWSZ + 255)/256, 256, 0, stream>>>(lutf, recon, comb);

    // ---- pack conv weights (BN folded) to bf16 [tap][n][k]
    k_pack<<<36, 256, 0, stream>>>((const float*)d_in[16], (const float*)d_in[17], nullptr, nullptr, 32, 3, 0, wpk(0), bpk(0)); // rp_cin
    k_pack<<<36, 256, 0, stream>>>((const float*)d_in[18], (const float*)d_in[19], (const float*)d_in[20], (const float*)d_in[21], 32, 32, 1, wpk(1), bpk(1)); // rp_rbA
    k_pack<<<36, 256, 0, stream>>>((const float*)d_in[22], (const float*)d_in[23], (const float*)d_in[24], (const float*)d_in[25], 32, 32, 1, wpk(2), bpk(2)); // rp_rbB
    k_pack<<<36, 256, 0, stream>>>((const float*)d_in[26], (const float*)d_in[27], nullptr, nullptr, 3, 32, 0, wpk(3), bpk(3)); // rp_cout
    k_pack<<<36, 256, 0, stream>>>((const float*)d_in[28], (const float*)d_in[29], nullptr, nullptr, 32, 6, 0, wpk(4), bpk(4)); // fu_cin
    k_pack<<<36, 256, 0, stream>>>((const float*)d_in[30], (const float*)d_in[31], (const float*)d_in[32], (const float*)d_in[33], 32, 32, 1, wpk(5), bpk(5)); // fu_rbA
    k_pack<<<36, 256, 0, stream>>>((const float*)d_in[34], (const float*)d_in[35], (const float*)d_in[36], (const float*)d_in[37], 32, 32, 1, wpk(6), bpk(6)); // fu_rbB
    k_pack<<<36, 256, 0, stream>>>((const float*)d_in[38], (const float*)d_in[39], nullptr, nullptr, 3, 32, 0, wpk(7), bpk(7)); // fu_cout

    // ---- WMMA conv pipeline: 36864 tiles, 4 waves/block -> 9216 blocks
    const int NBLK = BB * HH * (WWID/16) / 4;
    k_conv_wmma<<<NBLK, 128, 0, stream>>>(img, 3, 0, 3,  wpk(0), bpk(0), bufA, 32, 0, 32, nullptr, 0, 0, 1);
    k_conv_wmma<<<NBLK, 128, 0, stream>>>(bufA, 32, 0, 32, wpk(1), bpk(1), bufB, 32, 0, 32, nullptr, 0, 0, 1);
    k_conv_wmma<<<NBLK, 128, 0, stream>>>(bufB, 32, 0, 32, wpk(2), bpk(2), bufC, 32, 0, 32, bufA, 32, 0, 3);
    k_conv_wmma<<<NBLK, 128, 0, stream>>>(bufC, 32, 0, 32, wpk(3), bpk(3), comb, 6, 3, 3,  nullptr, 0, 0, 0);
    k_conv_wmma<<<NBLK, 128, 0, stream>>>(comb, 6, 0, 6,  wpk(4), bpk(4), bufA, 32, 0, 32, nullptr, 0, 0, 1);
    k_conv_wmma<<<NBLK, 128, 0, stream>>>(bufA, 32, 0, 32, wpk(5), bpk(5), bufB, 32, 0, 32, nullptr, 0, 0, 1);
    k_conv_wmma<<<NBLK, 128, 0, stream>>>(bufB, 32, 0, 32, wpk(6), bpk(6), bufC, 32, 0, 32, bufA, 32, 0, 3);
    k_conv_wmma<<<NBLK, 128, 0, stream>>>(bufC, 32, 0, 32, wpk(7), bpk(7), outp, 3, 0, 3,  img, 3, 0, 2);
}
